// TernaryLinear_68178310856945
// MI455X (gfx1250) — matrix operand order repaired and checked by reading, not compiled, not run on hardware
//
#include <hip/hip_runtime.h>
#include <hip/hip_bf16.h>

// Problem sizes (compile-time constants per the reference)
#define TOK    8192
#define INF    4096
#define OUTF   11008
#define NGRP   32      // INF / 128
#define BK     32      // K per LDS stage (one 16x16x32 WMMA step)
#define LDSS   40      // padded LDS row stride in bf16 elements (80B, 16B aligned)

typedef __bf16 bf16;
typedef bf16  v16bf __attribute__((ext_vector_type(16)));
typedef bf16  v8bf  __attribute__((ext_vector_type(8)));
typedef float v8f   __attribute__((ext_vector_type(8)));

// CDNA5 async memory->LDS copies (ASYNCcnt-tracked, no VGPR staging).
#if defined(__gfx1250__) && __has_builtin(__builtin_amdgcn_global_load_async_to_lds_b128)
#define HAVE_ASYNC_LDS 1
#else
#define HAVE_ASYNC_LDS 0
#endif

#if HAVE_ASYNC_LDS
// Per the compiler diagnostic, the builtin takes pointers to a 16-byte int
// vector: param0 in the global (__device__ / AS1) space, param1 in LDS (AS3).
typedef int v4i_vs __attribute__((vector_size(16)));
typedef __attribute__((address_space(1))) v4i_vs* g128_t;
typedef __attribute__((address_space(3))) v4i_vs* l128_t;

__device__ __forceinline__ void async_fill_b128(const bf16* g, bf16* l) {
  __builtin_amdgcn_global_load_async_to_lds_b128((g128_t)g, (l128_t)l, 0, 0);
}
__device__ __forceinline__ void wait_async0() {
#if __has_builtin(__builtin_amdgcn_s_wait_asynccnt)
  __builtin_amdgcn_s_wait_asynccnt(0);
#else
  asm volatile("s_wait_asynccnt 0x0" ::: "memory");
#endif
}
#endif

// ---------------------------------------------------------------------------
// Pass 1: x fp32 -> bf16 (8 elems / thread, b128 loads + b128 store)
// ---------------------------------------------------------------------------
__global__ __launch_bounds__(256) void convert_x_kernel(
    const float* __restrict__ x, bf16* __restrict__ xb) {
  size_t g = (size_t)blockIdx.x * 256 + threadIdx.x;
  const float4* p = (const float4*)x + g * 2;
  float4 a = p[0];
  float4 b = p[1];
  v8bf r;
  r[0] = (bf16)a.x; r[1] = (bf16)a.y; r[2] = (bf16)a.z; r[3] = (bf16)a.w;
  r[4] = (bf16)b.x; r[5] = (bf16)b.y; r[6] = (bf16)b.z; r[7] = (bf16)b.w;
  ((v8bf*)xb)[g] = r;
}

// ---------------------------------------------------------------------------
// Pass 2: W dequant: ternary fp32 * scale[o, i/128] -> bf16
// 8 contiguous elems / thread; a group (128) never straddles an 8-chunk.
// ---------------------------------------------------------------------------
__global__ __launch_bounds__(256) void dequant_w_kernel(
    const float* __restrict__ w, const float* __restrict__ scales,
    bf16* __restrict__ wb) {
  size_t g   = (size_t)blockIdx.x * 256 + threadIdx.x;
  size_t idx = g * 8;
  int o = (int)(idx >> 12);          // / INF (4096)
  int i = (int)(idx & (INF - 1));
  float s = scales[o * NGRP + (i >> 7)];
  const float4* p = (const float4*)(w + idx);
  float4 a = p[0];
  float4 b = p[1];
  v8bf r;
  r[0] = (bf16)(a.x * s); r[1] = (bf16)(a.y * s);
  r[2] = (bf16)(a.z * s); r[3] = (bf16)(a.w * s);
  r[4] = (bf16)(b.x * s); r[5] = (bf16)(b.y * s);
  r[6] = (bf16)(b.z * s); r[7] = (bf16)(b.w * s);
  ((v8bf*)wb)[g] = r;
}

// ---------------------------------------------------------------------------
// Pass 3: C[M,N] = A[M,K](bf16) * B[N,K]^T(bf16) + bias, f32 accum via WMMA.
// Block tile 128x128, BK=32, 8 waves (2x4), wave tile 64x32 (4x2 WMMA tiles).
// Double-buffered LDS; tile fills via GLOBAL_LOAD_ASYNC_TO_LDS_B128 when
// available (ASYNCcnt wait + one barrier per K stage).
// ---------------------------------------------------------------------------
__global__ __launch_bounds__(256) void ternary_gemm_kernel(
    const bf16* __restrict__ A,      // [TOK][INF]
    const bf16* __restrict__ B,      // [OUTF][INF]
    const float* __restrict__ bias,  // [OUTF]
    float* __restrict__ C) {         // [TOK][OUTF]
  __shared__ bf16 As[2][128 * LDSS];
  __shared__ bf16 Bs[2][128 * LDSS];

  const int tid     = threadIdx.x;
  const int lane    = tid & 31;
  const int waveId  = tid >> 5;
  const int waveM   = waveId >> 2;   // 0..1 -> 64-row slabs
  const int waveN   = waveId & 3;    // 0..3 -> 32-col slabs
  const int laneRow = lane & 15;
  const int laneHi  = lane >> 4;

  const int blockM = blockIdx.y * 128;
  const int blockN = blockIdx.x * 128;

  // Tile-fill coords: 128x32 bf16 = 512 chunks of 8; 2 chunks/thread.
  const int c0 = tid,        r0 = c0 >> 2, q0 = (c0 & 3) * 8;
  const int c1 = tid + 256,  r1 = c1 >> 2, q1 = (c1 & 3) * 8;

  const bf16* aG = A + (size_t)blockM * INF;
  const bf16* bG = B + (size_t)blockN * INF;

  v8f acc[4][2] = {};

  auto stage_load = [&](int s, int k0) {
#if HAVE_ASYNC_LDS
    async_fill_b128(aG + (size_t)r0 * INF + k0 + q0, &As[s][r0 * LDSS + q0]);
    async_fill_b128(aG + (size_t)r1 * INF + k0 + q1, &As[s][r1 * LDSS + q1]);
    async_fill_b128(bG + (size_t)r0 * INF + k0 + q0, &Bs[s][r0 * LDSS + q0]);
    async_fill_b128(bG + (size_t)r1 * INF + k0 + q1, &Bs[s][r1 * LDSS + q1]);
#else
    *(v8bf*)&As[s][r0 * LDSS + q0] = *(const v8bf*)(aG + (size_t)r0 * INF + k0 + q0);
    *(v8bf*)&As[s][r1 * LDSS + q1] = *(const v8bf*)(aG + (size_t)r1 * INF + k0 + q1);
    *(v8bf*)&Bs[s][r0 * LDSS + q0] = *(const v8bf*)(bG + (size_t)r0 * INF + k0 + q0);
    *(v8bf*)&Bs[s][r1 * LDSS + q1] = *(const v8bf*)(bG + (size_t)r1 * INF + k0 + q1);
#endif
  };

  stage_load(0, 0);
#if HAVE_ASYNC_LDS
  wait_async0();
#endif
  __syncthreads();

  const int KT = INF / BK;  // 128 stages
  for (int kt = 0; kt < KT; ++kt) {
    const int cur = kt & 1;
    if (kt + 1 < KT) stage_load(cur ^ 1, (kt + 1) * BK);

    // A fragments: 16-bit A 16x32 layout — lane = row, half-wave K base 0/8,
    // element j -> K = kb + j + (j & 8)  => chunks [kb, kb+7] and [kb+16, kb+23]
    v16bf afrag[4];
    const int kbA = laneHi * 8;
#pragma unroll
    for (int mi = 0; mi < 4; ++mi) {
      const bf16* p = &As[cur][(waveM * 64 + mi * 16 + laneRow) * LDSS + kbA];
      v8bf lo = *(const v8bf*)p;
      v8bf hi = *(const v8bf*)(p + 16);
      afrag[mi] = __builtin_shufflevector(lo, hi,
          0, 1, 2, 3, 4, 5, 6, 7, 8, 9, 10, 11, 12, 13, 14, 15);
    }

    // B fragments: B 32x16 layout — lane = col, same K order as A:
    // element j -> K = kb + j + (j & 8), kb = 0/8 per half-wave.
    v16bf bfrag[2];
    const int kbB = laneHi * 8;
#pragma unroll
    for (int ni = 0; ni < 2; ++ni) {
      const bf16* p = &Bs[cur][(waveN * 32 + ni * 16 + laneRow) * LDSS + kbB];
      v8bf lo = *(const v8bf*)p;
      v8bf hi = *(const v8bf*)(p + 16);
      bfrag[ni] = __builtin_shufflevector(lo, hi,
          0, 1, 2, 3, 4, 5, 6, 7, 8, 9, 10, 11, 12, 13, 14, 15);
    }

#pragma unroll
    for (int mi = 0; mi < 4; ++mi)
#pragma unroll
      for (int ni = 0; ni < 2; ++ni)
        acc[mi][ni] = __builtin_amdgcn_wmma_f32_16x16x32_bf16(
            false, afrag[mi], false, bfrag[ni],
            (short)0, acc[mi][ni], false, false);

#if HAVE_ASYNC_LDS
    wait_async0();   // producer's async fills for next stage are in LDS
#endif
    __syncthreads();
  }

  // Epilogue: C layout — lane 0-15 col=lane, VGPR r holds M=r (+8 for hi half).
#pragma unroll
  for (int ni = 0; ni < 2; ++ni) {
    const int col = blockN + waveN * 32 + ni * 16 + laneRow;
    const float bb = bias[col];
#pragma unroll
    for (int mi = 0; mi < 4; ++mi) {
      const int rowBase = blockM + waveM * 64 + mi * 16 + laneHi * 8;
#pragma unroll
      for (int r = 0; r < 8; ++r)
        C[(size_t)(rowBase + r) * OUTF + col] = acc[mi][ni][r] + bb;
    }
  }
}

// ---------------------------------------------------------------------------
extern "C" void kernel_launch(void* const* d_in, const int* in_sizes, int n_in,
                              void* d_out, int out_size, void* d_ws, size_t ws_size,
                              hipStream_t stream) {
  const float* x      = (const float*)d_in[0];
  const float* w      = (const float*)d_in[1];
  const float* scales = (const float*)d_in[2];
  const float* bias   = (const float*)d_in[3];
  float* out = (float*)d_out;

  // Workspace: bf16 x (67 MB) then bf16 dequant W (90 MB) => 157 MB, L2-resident.
  bf16* xb = (bf16*)d_ws;
  bf16* wb = xb + (size_t)TOK * INF;

  const int xblocks = (int)(((size_t)TOK * INF) / (8 * 256));   // 16384
  const int wblocks = (int)(((size_t)OUTF * INF) / (8 * 256));  // 22016
  convert_x_kernel<<<xblocks, 256, 0, stream>>>(x, xb);
  dequant_w_kernel<<<wblocks, 256, 0, stream>>>(w, scales, wb);

  dim3 grid(OUTF / 128, TOK / 128);  // (86, 64): N fastest => A panel reuse in L2
  ternary_gemm_kernel<<<grid, 256, 0, stream>>>(xb, wb, bias, out);
}
